// RGCN_26036091748511
// MI455X (gfx1250) — matrix operand order repaired and checked by reading, not compiled, not run on hardware
//
#include <hip/hip_runtime.h>
#include <hip/hip_bf16.h>
#include <math.h>

// ---------------- problem constants ----------------
#define NB    4
#define NN    8000
#define NT    32000          // NB*NN
#define FIN   256
#define FOUT  512
#define NE    640000
#define RR    44
#define PADE  (NE + RR * 64)         // relation buckets padded to 64
#define MAXT  (NE / 64 + RR)         // upper bound on edge tiles = 10044

typedef __bf16 bf16;
typedef __attribute__((ext_vector_type(16))) __bf16 v16bf;
typedef __attribute__((ext_vector_type(8)))  float  v8f;

// ---------------- CDNA5 async global->LDS staging ----------------
#if defined(__has_builtin)
#  if __has_builtin(__builtin_amdgcn_global_load_async_to_lds_b128)
#    define ASYNC_PATH 1
#  endif
#  if __has_builtin(__builtin_amdgcn_s_wait_asynccnt)
#    define HAVE_WAIT_ASYNC 1
#  endif
#endif
#ifndef ASYNC_PATH
#  define ASYNC_PATH 0
#endif

typedef int v4i __attribute__((vector_size(16)));
typedef __attribute__((address_space(1))) v4i* gv4i_ptr;   // global AS
typedef __attribute__((address_space(3))) v4i* lv4i_ptr;   // LDS AS

__device__ __forceinline__ void async_copy16(const bf16* g, bf16* l) {
#if ASYNC_PATH
  __builtin_amdgcn_global_load_async_to_lds_b128(
      (gv4i_ptr)(void*)(bf16*)g, (lv4i_ptr)(void*)l, 0, 0);
#else
  *(uint4*)l = *(const uint4*)g;
#endif
}

template<int N>
__device__ __forceinline__ void async_join() {
#if ASYNC_PATH
#  if defined(HAVE_WAIT_ASYNC)
  __builtin_amdgcn_s_wait_asynccnt(N);
#  else
  asm volatile("s_wait_asynccnt %0" :: "i"(N) : "memory");
#  endif
#endif
}

// split fp32 -> bf16 hi/lo pair (v ~= hi + lo, error ~2^-17 rel)
__device__ __forceinline__ void split2(float v, bf16& h, bf16& l) {
  h = (bf16)v;
  l = (bf16)(v - (float)h);
}

// ---------------- BN stats / finalize / BN+GELU(split) ----------------
template<int C>
__global__ __launch_bounds__(256) void bn_stats(const float* __restrict__ x,
                                                float* __restrict__ sums) {
  constexpr int PC = C / 256;
  float s[PC], q[PC];
#pragma unroll
  for (int i = 0; i < PC; ++i) { s[i] = 0.f; q[i] = 0.f; }
  const int r0 = blockIdx.x * (NT / 256);          // 125 rows per block
  for (int r = r0; r < r0 + (NT / 256); ++r) {
    const float* row = x + (size_t)r * C;
#pragma unroll
    for (int i = 0; i < PC; ++i) {
      float v = row[threadIdx.x + i * 256];
      s[i] += v; q[i] += v * v;
    }
  }
#pragma unroll
  for (int i = 0; i < PC; ++i) {
    atomicAdd(&sums[threadIdx.x + i * 256],     s[i]);
    atomicAdd(&sums[C + threadIdx.x + i * 256], q[i]);
  }
}

template<int C>
__global__ void bn_finalize(const float* __restrict__ sums,
                            const float* __restrict__ gamma,
                            const float* __restrict__ beta,
                            float* __restrict__ coeff) {
  int c = blockIdx.x * blockDim.x + threadIdx.x;
  if (c >= C) return;
  float mean = sums[c] * (1.0f / NT);
  float var  = sums[C + c] * (1.0f / NT) - mean * mean;
  float sc   = gamma[c] * rsqrtf(var + 1e-5f);
  coeff[c]     = sc;
  coeff[C + c] = beta[c] - mean * sc;
}

template<int C>
__global__ __launch_bounds__(256) void bn_gelu_split(const float* __restrict__ x,
                                                     const float* __restrict__ coeff,
                                                     bf16* __restrict__ hh,
                                                     bf16* __restrict__ hl) {
  size_t i = (size_t)blockIdx.x * 256 + threadIdx.x;
  int c = (int)(i & (C - 1));
  float v = x[i] * coeff[c] + coeff[C + c];
  float g = 0.5f * v * (1.0f + erff(v * 0.70710678118654752f));
  bf16 h, l; split2(g, h, l);
  hh[i] = h; hl[i] = l;
}

__global__ __launch_bounds__(256) void cvt_split(const float* __restrict__ x,
                                                 bf16* __restrict__ xh,
                                                 bf16* __restrict__ xl, size_t total) {
  size_t i = (size_t)blockIdx.x * 256 + threadIdx.x;
  if (i >= total) return;
  bf16 h, l; split2(x[i], h, l);
  xh[i] = h; xl[i] = l;
}

// weights [R][K][N] fp32 -> split bf16 transposed [R][N][K]
__global__ __launch_bounds__(256) void cvt_w_t(const float* __restrict__ src,
                                               bf16* __restrict__ dh,
                                               bf16* __restrict__ dl,
                                               int K, int N, size_t total) {
  size_t i = (size_t)blockIdx.x * 256 + threadIdx.x;
  if (i >= total) return;
  size_t rk = i / N; int n = (int)(i - rk * N);
  size_t r = rk / K; int k = (int)(rk - r * K);
  bf16 h, l; split2(src[i], h, l);
  size_t o = (r * N + n) * (size_t)K + k;
  dh[o] = h; dl[o] = l;
}

// ---------------- edge preprocessing ----------------
__global__ void count_edges(const int* __restrict__ ei, const int* __restrict__ ea,
                            int* __restrict__ cnt, int* __restrict__ relCnt) {
  int e = blockIdx.x * 256 + threadIdx.x;
  if (e >= NE) return;
  atomicAdd(&cnt[ei[NE + e] * RR + ea[e]], 1);
  atomicAdd(&relCnt[ea[e]], 1);
}

__global__ void setup_tiles(const int* __restrict__ relCnt, int* __restrict__ offA,
                            int* __restrict__ tileRel, int* __restrict__ tileBase,
                            int* __restrict__ nTiles) {
  if (threadIdx.x != 0 || blockIdx.x != 0) return;
  int off = 0, t = 0;
  for (int r = 0; r < RR; ++r) {
    offA[r] = off;
    int tl = (relCnt[r] + 63) >> 6;
    for (int i = 0; i < tl; ++i) { tileRel[t] = r; tileBase[t] = off + i * 64; ++t; }
    off += tl * 64;
  }
  offA[RR] = off;
  *nTiles  = t;
}

__global__ void scatter_edges(const int* __restrict__ ei, const int* __restrict__ ea,
                              const int* __restrict__ cnt, const int* __restrict__ offA,
                              int* __restrict__ cur, int* __restrict__ esrc,
                              int* __restrict__ edst, float* __restrict__ einv) {
  int e = blockIdx.x * 256 + threadIdx.x;
  if (e >= NE) return;
  int src = ei[e], dst = ei[NE + e], r = ea[e];
  int p = offA[r] + atomicAdd(&cur[r], 1);
  esrc[p] = src;
  edst[p] = dst;
  int c = cnt[dst * RR + r];
  einv[p] = 1.0f / (float)(c > 0 ? c : 1);
}

// ---------------- split-bf16 WMMA GEMM core ----------------
// Block tile BM=64 x BN=128, BK=32; 8 waves (4M x 2N); per chunk 12
// v_wmma_f32_16x16x32_bf16 (Ah*Bh + Ah*Bl + Al*Bh); double-buffered LDS with
// async global->LDS staging (6 async copies/thread/chunk).
#define BM 64
#define BN 128
#define BK 32
#define LSTR 56              // LDS row stride in bf16 (112B, conflict-free)

__device__ __forceinline__ v16bf load_fragA(const bf16* p) {
  v16bf r;                               // halves at +0 and +16
  ((uint4*)&r)[0] = *(const uint4*)p;
  ((uint4*)&r)[1] = *(const uint4*)(p + 16);
  return r;
}
__device__ __forceinline__ v16bf load_fragB(const bf16* p) {
  v16bf r;                               // halves at +0 and +16
  ((uint4*)&r)[0] = *(const uint4*)p;
  ((uint4*)&r)[1] = *(const uint4*)(p + 16);
  return r;
}
#define WMMA_BF16(A_, B_, C_) \
  __builtin_amdgcn_wmma_f32_16x16x32_bf16(false, (A_), false, (B_), (short)0, (C_), false, false)

// C[m,n] (=|+=atomic) sum_k (Ah+Al)[m,k]*(Bh+Bl)[n,k]^T + bias[n]
template<int K, bool ATOMIC>
__global__ __launch_bounds__(256) void dense_gemm_bf16(const bf16* __restrict__ Ahg,
                                                       const bf16* __restrict__ Alg,
                                                       const bf16* __restrict__ Bhg,
                                                       const bf16* __restrict__ Blg,
                                                       const float* __restrict__ bias,
                                                       float* __restrict__ C, int N) {
  __shared__ __align__(16) bf16 Ah[2][BM][LSTR], Al[2][BM][LSTR];
  __shared__ __align__(16) bf16 Bh[2][BN][LSTR], Bl[2][BN][LSTR];
  const int tid  = threadIdx.x;
  const int lane = tid & 31, wid = tid >> 5;
  const int wm = wid & 3, wn = wid >> 2;
  const int bm = blockIdx.x * BM, bn = blockIdx.y * BN;
  v8f acc[4] = {v8f{}, v8f{}, v8f{}, v8f{}};

  auto stage = [&](int buf, int kk) {
    {
      int row = tid >> 2, k8 = (tid & 3) << 3;
      size_t g = (size_t)(bm + row) * K + kk + k8;
      async_copy16(Ahg + g, &Ah[buf][row][k8]);
      async_copy16(Alg + g, &Al[buf][row][k8]);
    }
#pragma unroll
    for (int i = 0; i < 2; ++i) {
      int lin = tid + i * 256;
      int row = lin >> 2, k8 = (lin & 3) << 3;
      size_t g = (size_t)(bn + row) * K + kk + k8;
      async_copy16(Bhg + g, &Bh[buf][row][k8]);
      async_copy16(Blg + g, &Bl[buf][row][k8]);
    }
  };

  constexpr int NCH = K / BK;
  stage(0, 0);
  for (int c = 0; c < NCH; ++c) {
    const int buf = c & 1;
    if (c + 1 < NCH) { stage(buf ^ 1, (c + 1) * BK); async_join<6>(); }
    else             { async_join<0>(); }
    __syncthreads();
    {
      const int mr = wm * 16 + (lane & 15);
      const int ka = (lane >> 4) << 3;   // lanes0-15: K{0..7,16..23}; 16-31: {8..15,24..31}
      v16bf ah = load_fragA(&Ah[buf][mr][ka]);
      v16bf al = load_fragA(&Al[buf][mr][ka]);
      const int kb = (lane >> 4) << 3;   // lanes0-15: K{0..7,16..23}; 16-31: {8..15,24..31}
#pragma unroll
      for (int nt = 0; nt < 4; ++nt) {
        int nb = wn * 64 + nt * 16 + (lane & 15);
        v16bf bh = load_fragB(&Bh[buf][nb][kb]);
        v16bf bl = load_fragB(&Bl[buf][nb][kb]);
        acc[nt] = WMMA_BF16(ah, bh, acc[nt]);
        acc[nt] = WMMA_BF16(ah, bl, acc[nt]);
        acc[nt] = WMMA_BF16(al, bh, acc[nt]);
      }
    }
    __syncthreads();
  }
#pragma unroll
  for (int nt = 0; nt < 4; ++nt) {
    int col  = bn + wn * 64 + nt * 16 + (lane & 15);
    float bv = bias[col];
#pragma unroll
    for (int j = 0; j < 8; ++j) {
      int row = bm + wm * 16 + j + ((lane >> 4) << 3);
      float val = acc[nt][j] + bv;
      if (ATOMIC) atomicAdd(&C[(size_t)row * N + col], val);
      else        C[(size_t)row * N + col] = val;
    }
  }
}

// Edge-tile GEMM: 64 relation-sorted edges; A rows gathered (pads read from a
// zeroed scratch page so staging stays EXEC-uniform for the async DMA path);
// epilogue scales by 1/deg and scatter-atomicAdds into out[dst].
template<int K>
__global__ __launch_bounds__(256) void edge_gemm_bf16(const bf16* __restrict__ Hh,
                                                      const bf16* __restrict__ Hl,
                                                      const bf16* __restrict__ Whg,
                                                      const bf16* __restrict__ Wlg,
                                                      const bf16* __restrict__ zp,
                                                      const int* __restrict__ esrc,
                                                      const int* __restrict__ edst,
                                                      const float* __restrict__ einv,
                                                      const int* __restrict__ tileRel,
                                                      const int* __restrict__ tileBase,
                                                      const int* __restrict__ nTiles,
                                                      float* __restrict__ C, int N) {
  if ((int)blockIdx.x >= *nTiles) return;          // uniform early exit
  __shared__ __align__(16) bf16 Ah[2][BM][LSTR], Al[2][BM][LSTR];
  __shared__ __align__(16) bf16 Bh[2][BN][LSTR], Bl[2][BN][LSTR];
  __shared__ int   s_src[BM];
  __shared__ int   s_dst[BM];
  __shared__ float s_inv[BM];
  const int tid  = threadIdx.x;
  const int lane = tid & 31, wid = tid >> 5;
  const int wm = wid & 3, wn = wid >> 2;
  const int bn = blockIdx.y * BN;
  const int rel  = tileRel[blockIdx.x];
  const int base = tileBase[blockIdx.x];
  const bf16* Wh = Whg + (size_t)rel * K * N;      // [N][K] layout
  const bf16* Wl = Wlg + (size_t)rel * K * N;

  if (tid < BM) {
    int p = base + tid;
    int s = esrc[p];
    s_src[tid] = s;
    s_dst[tid] = (s >= 0) ? edst[p] : 0;
    s_inv[tid] = (s >= 0) ? einv[p] : 0.0f;
  }
  __syncthreads();

  v8f acc[4] = {v8f{}, v8f{}, v8f{}, v8f{}};

  auto stage = [&](int buf, int kk) {
    {
      int row = tid >> 2, k8 = (tid & 3) << 3;
      int s = s_src[row];
      const bf16* ph = (s >= 0) ? (Hh + (size_t)s * K + kk + k8) : zp;
      const bf16* pl = (s >= 0) ? (Hl + (size_t)s * K + kk + k8) : zp;
      async_copy16(ph, &Ah[buf][row][k8]);
      async_copy16(pl, &Al[buf][row][k8]);
    }
#pragma unroll
    for (int i = 0; i < 2; ++i) {
      int lin = tid + i * 256;
      int row = lin >> 2, k8 = (lin & 3) << 3;
      size_t g = (size_t)(bn + row) * K + kk + k8;
      async_copy16(Wh + g, &Bh[buf][row][k8]);
      async_copy16(Wl + g, &Bl[buf][row][k8]);
    }
  };

  constexpr int NCH = K / BK;
  stage(0, 0);
  for (int c = 0; c < NCH; ++c) {
    const int buf = c & 1;
    if (c + 1 < NCH) { stage(buf ^ 1, (c + 1) * BK); async_join<6>(); }
    else             { async_join<0>(); }
    __syncthreads();
    {
      const int mr = wm * 16 + (lane & 15);
      const int ka = (lane >> 4) << 3;
      v16bf ah = load_fragA(&Ah[buf][mr][ka]);
      v16bf al = load_fragA(&Al[buf][mr][ka]);
      const int kb = (lane >> 4) << 3;
#pragma unroll
      for (int nt = 0; nt < 4; ++nt) {
        int nb = wn * 64 + nt * 16 + (lane & 15);
        v16bf bh = load_fragB(&Bh[buf][nb][kb]);
        v16bf bl = load_fragB(&Bl[buf][nb][kb]);
        acc[nt] = WMMA_BF16(ah, bh, acc[nt]);
        acc[nt] = WMMA_BF16(ah, bl, acc[nt]);
        acc[nt] = WMMA_BF16(al, bh, acc[nt]);
      }
    }
    __syncthreads();
  }
#pragma unroll
  for (int nt = 0; nt < 4; ++nt) {
    int col = bn + wn * 64 + nt * 16 + (lane & 15);
#pragma unroll
    for (int j = 0; j < 8; ++j) {
      int m = wm * 16 + j + ((lane >> 4) << 3);
      if (s_src[m] >= 0)
        atomicAdd(&C[(size_t)s_dst[m] * N + col], acc[nt][j] * s_inv[m]);
    }
  }
}

// ---------------- launcher ----------------
extern "C" void kernel_launch(void* const* d_in, const int* in_sizes, int n_in,
                              void* d_out, int out_size, void* d_ws, size_t ws_size,
                              hipStream_t stream) {
  const float* x      = (const float*)d_in[0];
  const int*   ei     = (const int*)d_in[1];
  const int*   ea     = (const int*)d_in[2];
  const float* gamma1 = (const float*)d_in[3];
  const float* beta1  = (const float*)d_in[4];
  const float* W1     = (const float*)d_in[5];
  const float* root1  = (const float*)d_in[6];
  const float* b1     = (const float*)d_in[7];
  const float* gamma2 = (const float*)d_in[8];
  const float* beta2  = (const float*)d_in[9];
  const float* W2     = (const float*)d_in[10];
  const float* root2  = (const float*)d_in[11];
  const float* b2     = (const float*)d_in[12];
  const float* Wskip  = (const float*)d_in[13];
  const float* bskip  = (const float*)d_in[14];
  float* out = (float*)d_out;

  char* ws = (char*)d_ws;
  auto carve = [&](size_t bytes) -> char* {
    char* p = ws;
    ws += (bytes + 255) & ~(size_t)255;
    return p;
  };
  bf16* h1h  = (bf16*)carve((size_t)NT * FIN * 2);
  bf16* h1l  = (bf16*)carve((size_t)NT * FIN * 2);
  bf16* h2h  = (bf16*)carve((size_t)NT * FOUT * 2);
  bf16* h2l  = (bf16*)carve((size_t)NT * FOUT * 2);
  bf16* xh   = (bf16*)carve((size_t)NT * FIN * 2);
  bf16* xl   = (bf16*)carve((size_t)NT * FIN * 2);
  float* out1 = (float*)carve((size_t)NT * FOUT * 4);
  bf16* w1h  = (bf16*)carve((size_t)RR * FIN * FOUT * 2);
  bf16* w1l  = (bf16*)carve((size_t)RR * FIN * FOUT * 2);
  bf16* w2h  = (bf16*)carve((size_t)RR * FOUT * FOUT * 2);
  bf16* w2l  = (bf16*)carve((size_t)RR * FOUT * FOUT * 2);
  bf16* r1h  = (bf16*)carve((size_t)FIN * FOUT * 2);
  bf16* r1l  = (bf16*)carve((size_t)FIN * FOUT * 2);
  bf16* r2h  = (bf16*)carve((size_t)FOUT * FOUT * 2);
  bf16* r2l  = (bf16*)carve((size_t)FOUT * FOUT * 2);
  bf16* skh  = (bf16*)carve((size_t)FIN * FOUT * 2);
  bf16* skl  = (bf16*)carve((size_t)FIN * FOUT * 2);
  bf16* zp   = (bf16*)carve(256);          // zero page for pad-row DMA reads
  float* sums1  = (float*)carve(2 * FIN * 4);
  float* coeff1 = (float*)carve(2 * FIN * 4);
  float* sums2  = (float*)carve(2 * FOUT * 4);
  float* coeff2 = (float*)carve(2 * FOUT * 4);
  int*   cnt    = (int*)carve((size_t)NT * RR * 4);
  int*   relCnt = (int*)carve(RR * 4);
  int*   cur    = (int*)carve(RR * 4);
  int*   offA   = (int*)carve((RR + 1) * 4);
  int*   esrc   = (int*)carve((size_t)PADE * 4);
  int*   edst   = (int*)carve((size_t)PADE * 4);
  float* einv   = (float*)carve((size_t)PADE * 4);
  int*   tileRel  = (int*)carve(MAXT * 4);
  int*   tileBase = (int*)carve(MAXT * 4);
  int*   nTiles   = (int*)carve(4);

  (void)hipMemsetAsync(sums1, 0, 2 * FIN * 4, stream);
  (void)hipMemsetAsync(sums2, 0, 2 * FOUT * 4, stream);
  (void)hipMemsetAsync(cnt, 0, (size_t)NT * RR * 4, stream);
  (void)hipMemsetAsync(relCnt, 0, RR * 4, stream);
  (void)hipMemsetAsync(cur, 0, RR * 4, stream);
  (void)hipMemsetAsync(esrc, 0xFF, (size_t)PADE * 4, stream);   // pads = -1
  (void)hipMemsetAsync(edst, 0, (size_t)PADE * 4, stream);
  (void)hipMemsetAsync(einv, 0, (size_t)PADE * 4, stream);
  (void)hipMemsetAsync(zp, 0, 256, stream);

  // ---- weight / input conversions (split bf16, weights transposed) ----
  {
    size_t t1 = (size_t)RR * FIN * FOUT;
    cvt_w_t<<<(int)((t1 + 255) / 256), 256, 0, stream>>>(W1, w1h, w1l, FIN, FOUT, t1);
    size_t t2 = (size_t)RR * FOUT * FOUT;
    cvt_w_t<<<(int)((t2 + 255) / 256), 256, 0, stream>>>(W2, w2h, w2l, FOUT, FOUT, t2);
    size_t t3 = (size_t)FIN * FOUT;
    cvt_w_t<<<(int)((t3 + 255) / 256), 256, 0, stream>>>(root1, r1h, r1l, FIN, FOUT, t3);
    size_t t4 = (size_t)FOUT * FOUT;
    cvt_w_t<<<(int)((t4 + 255) / 256), 256, 0, stream>>>(root2, r2h, r2l, FOUT, FOUT, t4);
    cvt_w_t<<<(int)((t3 + 255) / 256), 256, 0, stream>>>(Wskip, skh, skl, FIN, FOUT, t3);
    size_t tx = (size_t)NT * FIN;
    cvt_split<<<(int)((tx + 255) / 256), 256, 0, stream>>>(x, xh, xl, tx);
  }

  // ---- layer 1: BN + GELU -> split bf16 ----
  bn_stats<FIN><<<256, 256, 0, stream>>>(x, sums1);
  bn_finalize<FIN><<<1, 256, 0, stream>>>(sums1, gamma1, beta1, coeff1);
  bn_gelu_split<FIN><<<(NT * FIN) / 256, 256, 0, stream>>>(x, coeff1, h1h, h1l);

  // ---- edge preprocessing (shared by both layers) ----
  count_edges<<<(NE + 255) / 256, 256, 0, stream>>>(ei, ea, cnt, relCnt);
  setup_tiles<<<1, 1, 0, stream>>>(relCnt, offA, tileRel, tileBase, nTiles);
  scatter_edges<<<(NE + 255) / 256, 256, 0, stream>>>(ei, ea, cnt, offA, cur,
                                                      esrc, edst, einv);

  // ---- layer 1 conv ----
  dim3 gd(NT / BM, FOUT / BN);
  dense_gemm_bf16<FIN, false><<<gd, 256, 0, stream>>>(h1h, h1l, r1h, r1l, b1, out1, FOUT);
  dim3 ge(MAXT, FOUT / BN);
  edge_gemm_bf16<FIN><<<ge, 256, 0, stream>>>(h1h, h1l, w1h, w1l, zp, esrc, edst, einv,
                                              tileRel, tileBase, nTiles, out1, FOUT);

  // ---- layer 2: BN + GELU -> split bf16 ----
  bn_stats<FOUT><<<256, 256, 0, stream>>>(out1, sums2);
  bn_finalize<FOUT><<<2, 256, 0, stream>>>(sums2, gamma2, beta2, coeff2);
  bn_gelu_split<FOUT><<<(NT * FOUT) / 256, 256, 0, stream>>>(out1, coeff2, h2h, h2l);

  // ---- layer 2 conv + skip into d_out ----
  dense_gemm_bf16<FOUT, false><<<gd, 256, 0, stream>>>(h2h, h2l, r2h, r2l, b2, out, FOUT);
  edge_gemm_bf16<FOUT><<<ge, 256, 0, stream>>>(h2h, h2l, w2h, w2l, zp, esrc, edst, einv,
                                               tileRel, tileBase, nTiles, out, FOUT);
  dense_gemm_bf16<FIN, true><<<gd, 256, 0, stream>>>(xh, xl, skh, skl, bskip, out, FOUT);
}
